// SPVCNN_backbone_76493367541912
// MI455X (gfx1250) — matrix operand order repaired and checked by reading, not compiled, not run on hardware
//
#include <hip/hip_runtime.h>
#include <hip/hip_bf16.h>
#include <stdint.h>

typedef _Float16 h16;
typedef __attribute__((ext_vector_type(16))) _Float16 v16h;
typedef __attribute__((ext_vector_type(8)))  _Float16 v8h;
typedef __attribute__((ext_vector_type(8)))  float    v8f;
typedef unsigned int u32;
typedef u32 u32x4 __attribute__((ext_vector_type(4)));
typedef int i32x4 __attribute__((ext_vector_type(4)));
typedef int i32x8 __attribute__((ext_vector_type(8)));

#define BN_EPS 1e-5f

// ---------------- elementwise / utility kernels ----------------

__global__ __launch_bounds__(256) void zero_k(float* p, long long n) {
  for (long long i = (long long)blockIdx.x * blockDim.x + threadIdx.x; i < n;
       i += (long long)gridDim.x * blockDim.x) p[i] = 0.f;
}

// scatter-mean voxelization (reads fp32 OR fp16 point features)
__global__ __launch_bounds__(256) void vox_scatter_k(
    const float* __restrict__ s32, const h16* __restrict__ s16, int sstride,
    const float* __restrict__ coords, int Np, int gs, float invs, int C,
    float* __restrict__ sums, float* __restrict__ cnt) {
  for (long long p = (long long)blockIdx.x * blockDim.x + threadIdx.x; p < Np;
       p += (long long)gridDim.x * blockDim.x) {
    float x = coords[p * 4 + 0] * invs;
    float y = coords[p * 4 + 1] * invs;
    float z = coords[p * 4 + 2] * invs;
    int i = min(max((int)floorf(x), 0), gs - 1);
    int j = min(max((int)floorf(y), 0), gs - 1);
    int k = min(max((int)floorf(z), 0), gs - 1);
    long long flat = ((long long)i * gs + j) * gs + k;
    atomicAdd(&cnt[flat], 1.f);
    for (int c = 0; c < C; ++c) {
      float v = s32 ? s32[p * (long long)sstride + c]
                    : (float)s16[p * (long long)sstride + c];
      atomicAdd(&sums[flat * C + c], v);
    }
  }
}

__global__ __launch_bounds__(256) void vox_finalize_k(
    const float* __restrict__ sums, const float* __restrict__ cnt,
    long long nvox, int C, int Cp, h16* __restrict__ g16,
    float* __restrict__ mask) {
  for (long long v = (long long)blockIdx.x * blockDim.x + threadIdx.x; v < nvox;
       v += (long long)gridDim.x * blockDim.x) {
    float n = cnt[v];
    if (mask) mask[v] = (n > 0.f) ? 1.f : 0.f;
    float inv = 1.f / fmaxf(n, 1.f);
    for (int c = 0; c < Cp; ++c)
      g16[v * Cp + c] = (c < C) ? (h16)(sums[v * C + c] * inv) : (h16)0.f;
  }
}

__global__ __launch_bounds__(256) void down_mask_k(
    const float* __restrict__ m, float* __restrict__ mo, int go) {
  long long n = (long long)go * go * go;
  int gi = go * 2;
  for (long long idx = (long long)blockIdx.x * blockDim.x + threadIdx.x; idx < n;
       idx += (long long)gridDim.x * blockDim.x) {
    int x = (int)(idx % go);
    long long t = idx / go;
    int y = (int)(t % go);
    int z = (int)(t / go);
    float v = 0.f;
    for (int dz = 0; dz < 2; ++dz)
      for (int dy = 0; dy < 2; ++dy)
        for (int dx = 0; dx < 2; ++dx)
          v = fmaxf(v, m[(((size_t)(2 * z + dz) * gi) + (2 * y + dy)) * gi +
                         (2 * x + dx)]);
    mo[idx] = v;
  }
}

// trilinear devoxelize: fp16 grid -> fp32 point features
__global__ __launch_bounds__(256) void devox_k(
    const h16* __restrict__ grid, const float* __restrict__ coords, int Np,
    int gs, float invs, int C, int Cp, float* __restrict__ out) {
  long long total = (long long)Np * C;
  for (long long idx = (long long)blockIdx.x * blockDim.x + threadIdx.x;
       idx < total; idx += (long long)gridDim.x * blockDim.x) {
    long long p = idx / C;
    int c = (int)(idx % C);
    float px = coords[p * 4 + 0] * invs;
    float py = coords[p * 4 + 1] * invs;
    float pz = coords[p * 4 + 2] * invs;
    float fx = floorf(px), fy = floorf(py), fz = floorf(pz);
    float rx = px - fx, ry = py - fy, rz = pz - fz;
    int ix = (int)fx, iy = (int)fy, iz = (int)fz;
    float acc = 0.f;
    for (int dx = 0; dx < 2; ++dx)
      for (int dy = 0; dy < 2; ++dy)
        for (int dz = 0; dz < 2; ++dz) {
          int cx = ix + dx, cy = iy + dy, cz = iz + dz;
          float w = (dx ? rx : 1.f - rx) * (dy ? ry : 1.f - ry) *
                    (dz ? rz : 1.f - rz);
          bool valid = (cx >= 0) && (cx < gs) && (cy >= 0) && (cy < gs) &&
                       (cz >= 0) && (cz < gs);
          int qx = min(max(cx, 0), gs - 1);
          int qy = min(max(cy, 0), gs - 1);
          int qz = min(max(cz, 0), gs - 1);
          float g = (float)grid[(((size_t)qx * gs + qy) * gs + qz) * Cp + c];
          acc += valid ? w * g : 0.f;
        }
    out[p * (long long)C + c] = acc;
  }
}

__global__ __launch_bounds__(256) void cvt16_k(
    const float* __restrict__ in, h16* __restrict__ out, long long nvox, int C,
    int Cp) {
  long long total = nvox * (long long)Cp;
  for (long long idx = (long long)blockIdx.x * blockDim.x + threadIdx.x;
       idx < total; idx += (long long)gridDim.x * blockDim.x) {
    long long v = idx / Cp;
    int c = (int)(idx % Cp);
    out[idx] = (c < C) ? (h16)in[v * (long long)C + c] : (h16)0.f;
  }
}

__global__ __launch_bounds__(256) void concat_k(
    const h16* __restrict__ a, int Ca, const h16* __restrict__ b, int Cb,
    h16* __restrict__ dst, long long nvox) {
  int Cp = Ca + Cb;
  long long total = nvox * (long long)Cp;
  for (long long idx = (long long)blockIdx.x * blockDim.x + threadIdx.x;
       idx < total; idx += (long long)gridDim.x * blockDim.x) {
    long long v = idx / Cp;
    int c = (int)(idx % Cp);
    dst[idx] = (c < Ca) ? a[v * (long long)Ca + c]
                        : b[v * (long long)Cb + (c - Ca)];
  }
}

// ---------------- BN over active voxels (masked) ----------------

__global__ __launch_bounds__(256) void bn_reduce_k(
    const float* __restrict__ x, const float* __restrict__ mask, long long nvox,
    int C, float* __restrict__ stats) {
  extern __shared__ float smem[];  // [0,C)=sum [C,2C)=sumsq [2C]=count
  for (int i = threadIdx.x; i < 2 * C + 1; i += blockDim.x) smem[i] = 0.f;
  __syncthreads();
  long long total = nvox * (long long)C;
  for (long long idx = (long long)blockIdx.x * blockDim.x + threadIdx.x;
       idx < total; idx += (long long)gridDim.x * blockDim.x) {
    long long v = idx / C;
    int c = (int)(idx % C);
    float mv = mask ? mask[v] : 1.f;
    float val = x[idx] * mv;
    atomicAdd(&smem[c], val);             // ds_add_f32
    atomicAdd(&smem[C + c], val * val);
    if (c == 0) atomicAdd(&smem[2 * C], mv);
  }
  __syncthreads();
  for (int i = threadIdx.x; i < 2 * C + 1; i += blockDim.x)
    if (smem[i] != 0.f) atomicAdd(&stats[i], smem[i]);
}

// flags: bit0 = relu, bit1 = add residual BEFORE relu (grid resblocks)
__global__ __launch_bounds__(256) void bn_apply_k(
    const float* __restrict__ x, const float* __restrict__ mask,
    const float* __restrict__ stats, const float* __restrict__ gam,
    const float* __restrict__ bet, long long nvox, int C, int Cp,
    h16* __restrict__ o16, float* __restrict__ o32,
    const h16* __restrict__ r16, const float* __restrict__ r32, int flags) {
  long long total = nvox * (long long)Cp;
  for (long long idx = (long long)blockIdx.x * blockDim.x + threadIdx.x;
       idx < total; idx += (long long)gridDim.x * blockDim.x) {
    long long v = idx / Cp;
    int c = (int)(idx % Cp);
    if (c >= C) { if (o16) o16[idx] = (h16)0.f; continue; }
    float cnt = fmaxf(stats[2 * C], 1.f);
    float mean = stats[c] / cnt;
    float var = stats[C + c] / cnt - mean * mean;
    float mv = mask ? mask[v] : 1.f;
    float y = (x[v * (long long)C + c] - mean) * rsqrtf(var + BN_EPS) * gam[c] +
              bet[c];
    y *= mv;
    float r = 0.f;
    if (r16) r = (float)r16[v * (long long)Cp + c];
    if (r32) r = r32[v * (long long)C + c];
    if (flags & 2) { y += r; r = 0.f; }
    if (flags & 1) y = fmaxf(y, 0.f);
    y += r;
    if (o16) o16[idx] = (h16)y;
    if (o32) o32[v * (long long)C + c] = y;
  }
}

// -------- repack weights (k,k,k,Cin,Cout) fp32 -> lane-major WMMA-B fp16 ----
// block layout: [(kt*ntiles + nt)*32 + lane]*16 + e ; lane = ((K%32>>3)&1)*16 + n%16, e = K%8 + (K%32>=16)*8
__global__ __launch_bounds__(256) void repack_w_k(
    const float* __restrict__ w, h16* __restrict__ wr, int ntaps, int Cin,
    int Cinp, int Cout) {
  int ntiles = Cout >> 4;
  long long total = (long long)ntaps * Cinp * Cout;
  for (long long idx = (long long)blockIdx.x * blockDim.x + threadIdx.x;
       idx < total; idx += (long long)gridDim.x * blockDim.x) {
    int n = (int)(idx % Cout);
    long long K = idx / Cout;
    int t = (int)(K / Cinp);
    int c = (int)(K % Cinp);
    float val = (c < Cin) ? w[((long long)t * Cin + c) * Cout + n] : 0.f;
    long long kt = K >> 5;
    int kin = (int)(K & 31);
    int lane = ((kin >> 3) & 1) * 16 + (n & 15);
    int e = (kin & 7) + ((kin >> 4) << 3);
    int nt = n >> 4;
    wr[((kt * ntiles + nt) * 32 + lane) * 16 + e] = (h16)val;
  }
}

// ---------------- implicit-GEMM conv via v_wmma_f32_16x16x32_f16 ----------
// NDHWC fp16 activations (Cinp mult of 32), Cout mult of 32.
// Each wave owns a 16-voxel strip and TWO adjacent 16-wide N-tiles.
// USE_LDS: whole repacked weight slab is DMA'd into LDS once per block by the
// Tensor Data Mover (wave 0 issues D#, waits TENSORcnt, block barrier), and
// B-fragments are then served from LDS (ds_load_b128).
template <bool USE_LDS>
__global__ __launch_bounds__(256) void conv_wmma_k(
    const h16* __restrict__ act, const h16* __restrict__ wrep,
    float* __restrict__ out, int gd, int gh, int gw, int od, int ohh, int oww_,
    int k, int s, int pad, int Cinp, int Cout, int slab8) {
  extern __shared__ h16 lds_w[];
  if constexpr (USE_LDS) {
    if (threadIdx.x < 32) {  // wave 0 of block issues the TDM transfer
      unsigned long long ga = (unsigned long long)(uintptr_t)wrep;
      u32 la = (u32)(uintptr_t)lds_w;
      // D# group0: count=1 (valid user desc), lds_addr, global_addr, type=2
      u32x4 g0 = {1u, la, (u32)ga,
                  (u32)(((ga >> 32) & 0x1FFFFFFu) | (2u << 30))};
      // D# group1: data_size=8B, tensor_dim0=tile_dim0=slab8, dim1=1, mask=0
      i32x8 g1 = {(int)(3u << 16),
                  (int)(((u32)slab8 & 0xFFFFu) << 16),
                  (int)((((u32)slab8 >> 16) & 0xFFFFu) | (1u << 16)),
                  (int)(((u32)slab8 & 0xFFFFu) << 16),
                  1,
                  slab8,
                  0,
                  0};
      i32x4 gz = {0, 0, 0, 0};
      (void)gz;
#if __has_builtin(__builtin_amdgcn_tensor_load_to_lds)
#if defined(__clang_major__) && (__clang_major__ >= 23)
      i32x8 gz8 = {0, 0, 0, 0, 0, 0, 0, 0};
      __builtin_amdgcn_tensor_load_to_lds(g0, g1, gz, gz, gz8, 0);
#else
      __builtin_amdgcn_tensor_load_to_lds(g0, g1, gz, gz, 0);
#endif
      __builtin_amdgcn_s_wait_tensorcnt(0);
#else
      (void)g0; (void)g1;
      const unsigned long long* src = (const unsigned long long*)wrep;
      unsigned long long* dst = (unsigned long long*)lds_w;
      for (int i = threadIdx.x; i < slab8; i += 32) dst[i] = src[i];
#endif
    }
    __syncthreads();
  }
  int ntiles = Cout >> 4;
  int npair = Cout >> 5;
  int wtiles = (oww_ + 15) >> 4;
  long long wave = ((long long)blockIdx.x * blockDim.x + threadIdx.x) >> 5;
  long long tasks = (long long)od * ohh * wtiles * npair;
  if (wave >= tasks) return;          // wave-uniform: EXEC stays all-ones
  int np = (int)(wave % npair);
  long long t1 = wave / npair;
  int wt = (int)(t1 % wtiles); t1 /= wtiles;
  int oy = (int)(t1 % ohh);
  int oz = (int)(t1 / ohh);
  int nt0 = np * 2;
  int lane = threadIdx.x & 31;
  int m = lane & 15, hs = lane >> 4, khalf = hs * 8;
  int ow = wt * 16 + m;
  bool vm = ow < oww_;
  int kk = k * k, ntaps = k * kk, nkc = Cinp >> 5;
  v8f acc0 = {}, acc1 = {};
  for (int t = 0; t < ntaps; ++t) {
    int tz = t / kk, ty = (t / k) % k, tx = t % k;
    int iz = oz * s + tz - pad, iy = oy * s + ty - pad, ix = ow * s + tx - pad;
    bool inb = vm && (unsigned)iz < (unsigned)gd && (unsigned)iy < (unsigned)gh &&
               (unsigned)ix < (unsigned)gw;
    const h16* abase =
        inb ? act + (((size_t)iz * gh + iy) * gw + ix) * Cinp + khalf : nullptr;
    size_t boff = (((size_t)t * nkc * ntiles + nt0) * 32 + lane) * 16;
    for (int kc = 0; kc < nkc; ++kc) {
      v16h a = {};
      if (inb) {
        v8h lo = *(const v8h*)(abase);        // K = khalf..khalf+7
        v8h hi = *(const v8h*)(abase + 16);   // K = 16+khalf..+7
        a = __builtin_shufflevector(lo, hi, 0, 1, 2, 3, 4, 5, 6, 7, 8, 9, 10,
                                    11, 12, 13, 14, 15);
        abase += 32;
      }
      v16h b0, b1;
      if constexpr (USE_LDS) {
        b0 = *(const v16h*)(lds_w + boff);
        b1 = *(const v16h*)(lds_w + boff + 32 * 16);
      } else {
        b0 = *(const v16h*)(wrep + boff);
        b1 = *(const v16h*)(wrep + boff + 32 * 16);
        __builtin_prefetch(wrep + boff + (size_t)ntiles * 32 * 16, 0, 0);
      }
      acc0 = __builtin_amdgcn_wmma_f32_16x16x32_f16(false, a, false, b0,
                                                    (short)0, acc0, false, false);
      acc1 = __builtin_amdgcn_wmma_f32_16x16x32_f16(false, a, false, b1,
                                                    (short)0, acc1, false, false);
      boff += (size_t)ntiles * 32 * 16;
    }
  }
  int col = nt0 * 16 + (lane & 15);
#pragma unroll
  for (int r = 0; r < 8; ++r) {
    int mm = r + hs * 8;
    int o = wt * 16 + mm;
    if (o < oww_) {
      size_t base = (((size_t)oz * ohh + oy) * oww_ + o) * Cout + col;
      out[base] = acc0[r];
      out[base + 16] = acc1[r];
    }
  }
}

// stride-2 k=2 transposed conv, decomposed by output parity -> 8 pointwise GEMMs
__global__ __launch_bounds__(256) void deconv_wmma_k(
    const h16* __restrict__ act, const h16* __restrict__ wrep,
    float* __restrict__ out, int gc, int Cinp, int Cout) {
  int ntiles = Cout >> 4;
  int npair = Cout >> 5;
  int wtiles = (gc + 15) >> 4;
  long long wave = ((long long)blockIdx.x * blockDim.x + threadIdx.x) >> 5;
  long long tasks = (long long)8 * gc * gc * wtiles * npair;
  if (wave >= tasks) return;
  int np = (int)(wave % npair);
  long long t1 = wave / npair;
  int wt = (int)(t1 % wtiles); t1 /= wtiles;
  int oy = (int)(t1 % gc); t1 /= gc;
  int oz = (int)(t1 % gc);
  int par = (int)(t1 / gc);
  int pz = par >> 2, py = (par >> 1) & 1, px = par & 1;
  int nt0 = np * 2;
  int lane = threadIdx.x & 31;
  int m = lane & 15, hs = lane >> 4, khalf = hs * 8;
  int w = wt * 16 + m;
  bool vm = w < gc;
  int nkc = Cinp >> 5;
  v8f acc0 = {}, acc1 = {};
  const h16* abase =
      vm ? act + (((size_t)oz * gc + oy) * gc + w) * Cinp + khalf : nullptr;
  const h16* bbase =
      wrep + (((long long)par * nkc * ntiles + nt0) * 32 + lane) * 16;
  for (int kc = 0; kc < nkc; ++kc) {
    v16h a = {};
    if (vm) {
      v8h lo = *(const v8h*)(abase);
      v8h hi = *(const v8h*)(abase + 16);
      a = __builtin_shufflevector(lo, hi, 0, 1, 2, 3, 4, 5, 6, 7, 8, 9, 10, 11,
                                  12, 13, 14, 15);
      abase += 32;
    }
    v16h b0 = *(const v16h*)(bbase);
    v16h b1 = *(const v16h*)(bbase + 32 * 16);
    acc0 = __builtin_amdgcn_wmma_f32_16x16x32_f16(false, a, false, b0, (short)0,
                                                  acc0, false, false);
    acc1 = __builtin_amdgcn_wmma_f32_16x16x32_f16(false, a, false, b1, (short)0,
                                                  acc1, false, false);
    bbase += (size_t)ntiles * 32 * 16;
  }
  int gf = gc * 2;
  int col = nt0 * 16 + (lane & 15);
#pragma unroll
  for (int r = 0; r < 8; ++r) {
    int mm = r + hs * 8;
    int ww = wt * 16 + mm;
    if (ww < gc) {
      size_t base = (((size_t)(2 * oz + pz) * gf + (2 * oy + py)) * gf +
                     (2 * ww + px)) * Cout + col;
      out[base] = acc0[r];
      out[base + 16] = acc1[r];
    }
  }
}

// ------------------------------- host side --------------------------------

extern "C" void kernel_launch(void* const* d_in, const int* in_sizes, int n_in,
                              void* d_out, int out_size, void* d_ws,
                              size_t ws_size, hipStream_t stream) {
  const float* feats = (const float*)d_in[0];
  const float* coords = (const float*)d_in[1];
  const int NP = in_sizes[0] / 4;

  // ---- parameter fetch (JAX pytree flattening = sorted dict keys) ----
  int pi = 2;
  auto F = [&]() { return (const float*)d_in[pi++]; };
  struct CB { const float *b, *g, *w; };
  struct RB { const float *b1, *g1, *b2, *g2, *db, *dg, *dw, *w1, *w2; bool has_dw; };
  auto getCB = [&]() { CB c; c.b = F(); c.g = F(); c.w = F(); return c; };
  auto getRB = [&](bool dw) {
    RB r; r.has_dw = dw;
    r.b1 = F(); r.g1 = F(); r.b2 = F(); r.g2 = F();
    if (dw) { r.db = F(); r.dg = F(); r.dw = F(); }
    else { r.db = r.dg = r.dw = nullptr; }
    r.w1 = F(); r.w2 = F(); return r;
  };
  // sorted: pt0, pt1, stage1..4, stem, up1..3
  const float *pt0_b = F(), *pt0_g = F(), *pt0_lb = F(), *pt0_lw = F();
  const float *pt1_b = F(), *pt1_g = F(), *pt1_lb = F(), *pt1_lw = F();
  (void)pt0_lb; (void)pt1_lb;  // bias before BN1d cancels exactly
  CB s1d = getCB(); RB s1r1 = getRB(false), s1r2 = getRB(false);
  CB s2d = getCB(); RB s2r1 = getRB(true),  s2r2 = getRB(false);
  CB s3d = getCB(); RB s3r1 = getRB(true),  s3r2 = getRB(false);
  CB s4d = getCB(); RB s4r1 = getRB(true),  s4r2 = getRB(false);
  const float *st_b1 = F(), *st_g1 = F(), *st_b2 = F(), *st_g2 = F(),
              *st_w1 = F(), *st_w2 = F();
  CB u1d = getCB(); RB u1r1 = getRB(true), u1r2 = getRB(false);
  CB u2d = getCB(); RB u2r1 = getRB(true), u2r2 = getRB(false);
  CB u3d = getCB(); RB u3r1 = getRB(true), u3r2 = getRB(false);

  // ---- workspace bump allocator ----
  char* wp = (char*)d_ws;
  auto A = [&](size_t bytes) -> void* {
    void* p = wp; wp += (bytes + 255) & ~(size_t)255; return p;
  };
  const long long nv0 = 262144, nv1 = 32768, nv2 = 4096, nv3 = 512, nv4 = 64;
  float *m0 = (float*)A(nv0 * 4), *m1 = (float*)A(nv1 * 4),
        *m2 = (float*)A(nv2 * 4), *m3 = (float*)A(nv3 * 4),
        *m4 = (float*)A(nv4 * 4);
  float* stats = (float*)A((2 * 384 + 1) * 4);
  h16* wrep = (h16*)A((size_t)27 * 384 * 256 * 2);
  float* cbuf = (float*)A((size_t)NP * 256 * 4);  // conv/linear fp32 out
  float* pbuf = (float*)A((size_t)NP * 256 * 4);  // devoxelize fp32 out
  float* vsum = (float*)A((size_t)nv0 * 32 * 4);
  float* vcnt = (float*)A((size_t)nv0 * 4);
  h16* z0_16 = (h16*)A((size_t)NP * 32 * 2);
  h16* z1_16 = (h16*)A((size_t)NP * 256 * 2);
  h16* x0f = (h16*)A((size_t)nv0 * 32 * 2);
  h16* x1f = (h16*)A((size_t)nv1 * 32 * 2);
  h16* x2f = (h16*)A((size_t)nv2 * 64 * 2);
  h16* x3f = (h16*)A((size_t)nv3 * 128 * 2);
  h16* x4f = (h16*)A((size_t)nv4 * 256 * 2);
  h16* T0 = (h16*)A((size_t)nv0 * 32 * 2);
  h16* T1 = (h16*)A((size_t)nv0 * 32 * 2);
  h16* T2 = (h16*)A((size_t)nv0 * 32 * 2);
  h16* T3 = (h16*)A((size_t)nv0 * 32 * 2);

  float* out_y3 = (float*)d_out;                      // (32,32,32,96)
  float* out_z2 = out_y3 + (long long)nv1 * 96;       // (NP,128)
  float* out_x0f = out_z2 + (long long)NP * 128;      // (64,64,64,32)

  auto NB = [](long long n) -> int {
    long long b = (n + 255) / 256;
    if (b < 1) b = 1; if (b > 1048576) b = 1048576; return (int)b;
  };
  auto Z = [&](float* p, long long n) { zero_k<<<NB(n), 256, 0, stream>>>(p, n); };

  auto CONV = [&](const h16* in, const float* w, int k, int s, int pad, int gd,
                  int gh, int gw, int od, int oh, int ow, int Cin, int Cinp,
                  int Cout, float* out) {
    long long rp = (long long)k * k * k * Cinp * Cout;
    repack_w_k<<<NB(rp), 256, 0, stream>>>(w, wrep, k * k * k, Cin, Cinp, Cout);
    long long thr = (long long)od * oh * ((ow + 15) / 16) * (Cout / 32) * 32;
    int blocks = (int)((thr + 255) / 256);
    long long slabBytes = (long long)k * k * k * Cinp * Cout * 2;
    int slab8 = (int)(slabBytes / 8);
    if (slabBytes <= 65536)   // weight slab fits in LDS: TDM-staged B path
      conv_wmma_k<true><<<blocks, 256, (size_t)slabBytes, stream>>>(
          in, wrep, out, gd, gh, gw, od, oh, ow, k, s, pad, Cinp, Cout, slab8);
    else
      conv_wmma_k<false><<<blocks, 256, 0, stream>>>(
          in, wrep, out, gd, gh, gw, od, oh, ow, k, s, pad, Cinp, Cout, slab8);
  };
  auto BN = [&](const float* x, const float* mk, const float* gam,
                const float* bet, long long nvox, int C, int Cp, h16* o16,
                float* o32, const h16* r16, const float* r32, int flags) {
    Z(stats, 2 * 384 + 1);
    int blocks = NB(nvox * (long long)C); if (blocks > 2048) blocks = 2048;
    bn_reduce_k<<<blocks, 256, (2 * C + 1) * sizeof(float), stream>>>(
        x, mk, nvox, C, stats);
    bn_apply_k<<<NB(nvox * (long long)Cp), 256, 0, stream>>>(
        x, mk, stats, gam, bet, nvox, C, Cp, o16, o32, r16, r32, flags);
  };
  auto RES = [&](const h16* in, int g, long long nvox, const float* mk,
                 const RB& P, int Cin, int Cout, h16* out, h16* tmp,
                 float* o32) {
    CONV(in, P.w1, 3, 1, 1, g, g, g, g, g, g, Cin, Cin, Cout, cbuf);
    BN(cbuf, mk, P.g1, P.b1, nvox, Cout, Cout, tmp, nullptr, nullptr, nullptr, 1);
    const h16* sc = in;
    if (P.has_dw) {
      CONV(in, P.dw, 1, 1, 0, g, g, g, g, g, g, Cin, Cin, Cout, cbuf);
      BN(cbuf, mk, P.dg, P.db, nvox, Cout, Cout, out, nullptr, nullptr, nullptr, 0);
      sc = out;
    }
    CONV(tmp, P.w2, 3, 1, 1, g, g, g, g, g, g, Cout, Cout, Cout, cbuf);
    BN(cbuf, mk, P.g2, P.b2, nvox, Cout, Cout, out, o32, sc, nullptr, 3);
  };
  auto DOWN = [&](const h16* in, int gin, const float* mi, float* mo,
                  const CB& P, int C, h16* out) {
    int go = gin / 2;
    long long nvo = (long long)go * go * go;
    down_mask_k<<<NB(nvo), 256, 0, stream>>>(mi, mo, go);
    CONV(in, P.w, 2, 2, 0, gin, gin, gin, go, go, go, C, C, C, cbuf);
    BN(cbuf, mo, P.g, P.b, nvo, C, C, out, nullptr, nullptr, nullptr, 1);
  };
  auto DEBR = [&](const h16* in, int gc, const float* mt, const CB& P, int Cin,
                  int Cout, h16* out) {
    long long rp = (long long)8 * Cin * Cout;
    repack_w_k<<<NB(rp), 256, 0, stream>>>(P.w, wrep, 8, Cin, Cin, Cout);
    long long thr =
        (long long)8 * gc * gc * ((gc + 15) / 16) * (Cout / 32) * 32;
    deconv_wmma_k<<<(int)((thr + 255) / 256), 256, 0, stream>>>(
        in, wrep, cbuf, gc, Cin, Cout);
    int gf = gc * 2;
    long long nvf = (long long)gf * gf * gf;
    BN(cbuf, mt, P.g, P.b, nvf, Cout, Cout, out, nullptr, nullptr, nullptr, 1);
  };
  auto VOX = [&](const float* s32, const h16* s16, int Csrc, int gs, float invs,
                 int Cp, h16* g16, float* mk) {
    long long nvox = (long long)gs * gs * gs;
    Z(vsum, nvox * Csrc);
    Z(vcnt, nvox);
    vox_scatter_k<<<NB(NP), 256, 0, stream>>>(s32, s16, Csrc, coords, NP, gs,
                                              invs, Csrc, vsum, vcnt);
    vox_finalize_k<<<NB(nvox), 256, 0, stream>>>(vsum, vcnt, nvox, Csrc, Cp,
                                                 g16, mk);
  };

  // ---------------- network ----------------
  // initial voxelize: feats (C=4, padded to 32) -> T0, m0
  VOX(feats, nullptr, 4, 64, 1.f, 32, T0, m0);
  // stem
  CONV(T0, st_w1, 3, 1, 1, 64, 64, 64, 64, 64, 64, 4, 32, 32, cbuf);
  BN(cbuf, m0, st_g1, st_b1, nv0, 32, 32, T1, nullptr, nullptr, nullptr, 1);
  CONV(T1, st_w2, 3, 1, 1, 64, 64, 64, 64, 64, 64, 32, 32, 32, cbuf);
  BN(cbuf, m0, st_g2, st_b2, nv0, 32, 32, x0f, out_x0f, nullptr, nullptr, 1);
  // z0 = devox(x0f, s=1); x1g = voxelize(z0)
  devox_k<<<NB((long long)NP * 32), 256, 0, stream>>>(x0f, coords, NP, 64, 1.f,
                                                      32, 32, pbuf);
  cvt16_k<<<NB((long long)NP * 32), 256, 0, stream>>>(pbuf, z0_16, NP, 32, 32);
  VOX(pbuf, nullptr, 32, 64, 1.f, 32, T0, nullptr);
  // stages 1-4
  DOWN(T0, 64, m0, m1, s1d, 32, T1);
  RES(T1, 32, nv1, m1, s1r1, 32, 32, T2, T3, nullptr);
  RES(T2, 32, nv1, m1, s1r2, 32, 32, x1f, T3, nullptr);
  DOWN(x1f, 32, m1, m2, s2d, 32, T1);
  RES(T1, 16, nv2, m2, s2r1, 32, 64, T2, T3, nullptr);
  RES(T2, 16, nv2, m2, s2r2, 64, 64, x2f, T3, nullptr);
  DOWN(x2f, 16, m2, m3, s3d, 64, T1);
  RES(T1, 8, nv3, m3, s3r1, 64, 128, T2, T3, nullptr);
  RES(T2, 8, nv3, m3, s3r2, 128, 128, x3f, T3, nullptr);
  DOWN(x3f, 8, m3, m4, s4d, 128, T1);
  RES(T1, 4, nv4, m4, s4r1, 128, 256, T2, T3, nullptr);
  RES(T2, 4, nv4, m4, s4r2, 256, 256, x4f, T3, nullptr);
  // point fusion 1: z1 = devox(x4f,16) + relu(bn1d(z0 @ pt0_lw))
  devox_k<<<NB((long long)NP * 256), 256, 0, stream>>>(x4f, coords, NP, 4,
                                                       1.f / 16.f, 256, 256, pbuf);
  CONV(z0_16, pt0_lw, 1, 1, 0, 1, 1, NP, 1, 1, NP, 32, 32, 256, cbuf);
  BN(cbuf, nullptr, pt0_g, pt0_b, NP, 256, 256, z1_16, nullptr, nullptr, pbuf, 1);
  VOX(nullptr, z1_16, 256, 4, 1.f / 16.f, 256, T0, nullptr);
  // up1
  DEBR(T0, 4, m3, u1d, 256, 256, T1);
  concat_k<<<NB(nv3 * 384), 256, 0, stream>>>(T1, 256, x3f, 128, T2, nv3);
  RES(T2, 8, nv3, m3, u1r1, 384, 256, T1, T3, nullptr);
  RES(T1, 8, nv3, m3, u1r2, 256, 256, T2, T3, nullptr);  // y1 -> T2
  // up2
  DEBR(T2, 8, m2, u2d, 256, 128, T1);
  concat_k<<<NB(nv2 * 192), 256, 0, stream>>>(T1, 128, x2f, 64, T3, nv2);
  RES(T3, 16, nv2, m2, u2r1, 192, 128, T1, T0, nullptr);
  RES(T1, 16, nv2, m2, u2r2, 128, 128, T2, T0, nullptr);  // y2 -> T2
  // point fusion 2: z2 = devox(y2,4) + relu(bn1d(z1 @ pt1_lw)) -> d_out
  devox_k<<<NB((long long)NP * 128), 256, 0, stream>>>(T2, coords, NP, 16,
                                                       0.25f, 128, 128, pbuf);
  CONV(z1_16, pt1_lw, 1, 1, 0, 1, 1, NP, 1, 1, NP, 256, 256, 128, cbuf);
  BN(cbuf, nullptr, pt1_g, pt1_b, NP, 128, 128, nullptr, out_z2, nullptr, pbuf, 1);
  VOX(out_z2, nullptr, 128, 16, 0.25f, 128, T0, nullptr);
  // up3 -> y3 (fp32 to d_out)
  DEBR(T0, 16, m1, u3d, 128, 96, T1);
  concat_k<<<NB(nv1 * 128), 256, 0, stream>>>(T1, 96, x1f, 32, T3, nv1);
  RES(T3, 32, nv1, m1, u3r1, 128, 96, T1, T0, nullptr);
  RES(T1, 32, nv1, m1, u3r2, 96, 96, T3, T0, out_y3);
}
